// MTP_5093831213509
// MI455X (gfx1250) — matrix operand order repaired and checked by reading, not compiled, not run on hardware
//
#include <hip/hip_runtime.h>
#include <hip/hip_bf16.h>

// ---------------------------------------------------------------------------
// MTP transformer: full layer + K=4 MTP heads.
// GEMMs: bf16 WMMA (f32 accum); weights pre-converted to bf16 once per launch;
// weight tiles staged to LDS with gfx1250 async DMA (global_load_async_to_lds).
// B=4 S=128 D=768 H=12 DK=64 V=32000 K=4, M=S-K=124.
// ---------------------------------------------------------------------------

#define BB 4
#define SS 128
#define DD 768
#define HH 12
#define DKH 64
#define VV 32000
#define KK 4
#define MM (SS - KK)      // 124
#define MROWS (BB * MM)   // 496

typedef __attribute__((ext_vector_type(16))) __bf16 v16bf;
typedef __attribute__((ext_vector_type(8)))  float  v8f;

// ---------------- gfx1250 async-LDS helpers ---------------------------------
__device__ __forceinline__ void async_copy_b128(unsigned lds_addr, const void* gaddr) {
    // GLOBAL_LOAD_ASYNC_TO_LDS_B128: per-lane 16B global -> LDS, ASYNCcnt-tracked.
    asm volatile("global_load_async_to_lds_b128 %0, %1, off"
                 :: "v"(lds_addr), "v"(gaddr) : "memory");
}
__device__ __forceinline__ void wait_async0() {
#if __has_builtin(__builtin_amdgcn_s_wait_asynccnt)
    __builtin_amdgcn_s_wait_asynccnt(0);
#else
    asm volatile("s_wait_asynccnt 0x0" ::: "memory");
#endif
}

// ---------------- WMMA GEMM: C[M,N] = act(A[M,Kd] @ B[Kd,N] + bias) ---------
// Block: 256 threads (8 waves), tile 64(M) x 256(N), K-step 64.
// Wave w: wm = w>>2, wn = w&3; wave tile 32x64 = 2x4 v8f accumulators
// -> 16 v_wmma per barrier window. A (fp32 activations) converted in-kernel;
// B (bf16 weights) staged via async DMA.

#define TM 64
#define TN 256
#define TKS 64                 // K elements staged per barrier
#define ASTR 72                // A LDS row stride (bf16): 144B (16B-mult, banks spread)
#define BSTR2 264              // B LDS row stride (bf16): 528B (16B-mult, 2 hits/bank)

__device__ __forceinline__ v16bf load_a_frag(const __bf16* As, int mbase, int ko, int lane) {
    // 16-bit A 16x32 layout (ISA 7.12.2): lanes 0-15 -> M=lane, K {0..7,16..23};
    // lanes 16-31 -> same M, K {8..15,24..31}. Each half is 8 contiguous bf16.
    v16bf f;
    const int m  = mbase + (lane & 15);
    const int kh = ko + ((lane >> 4) << 3);
    const __bf16* p0 = As + m * ASTR + kh;        // K = kh .. kh+7
    const __bf16* p1 = p0 + 16;                   // K = kh+16 .. kh+23
#pragma unroll
    for (int i = 0; i < 8; ++i) f[i] = p0[i];
#pragma unroll
    for (int i = 0; i < 8; ++i) f[8 + i] = p1[i];
    return f;
}

__device__ __forceinline__ v16bf load_b_frag(const __bf16* Bs, int ko, int nbase, int lane) {
    // 16-bit B 32x16: lane L holds column N=L&15, K {0..7,16..23} (+8 for lanes 16-31).
    v16bf f;
    const __bf16* col = Bs + (ko + ((lane >> 4) << 3)) * BSTR2 + nbase + (lane & 15);
#pragma unroll
    for (int i = 0; i < 8; ++i) f[i] = col[i * BSTR2];
#pragma unroll
    for (int i = 0; i < 8; ++i) f[8 + i] = col[(16 + i) * BSTR2];
    return f;
}

__global__ __launch_bounds__(256) void gemm_bf16_wmma(
    const float*  __restrict__ A, int lda,
    const __bf16* __restrict__ Bm, int ldb,
    const float*  __restrict__ bias,
    float* __restrict__ C, long long ldc,
    int M, int Kd, int relu)
{
    __shared__ __bf16 As[TM * ASTR];       //  9216 B
    __shared__ __bf16 Bs[TKS * BSTR2];     // 33792 B

    const int tid  = threadIdx.x;
    const int lane = tid & 31;
    const int wave = tid >> 5;
    const int wm   = wave >> 2;   // 0..1
    const int wn   = wave & 3;    // 0..3

    const int n0 = blockIdx.x * TN;
    const int m0 = blockIdx.y * TM;

    v8f acc[2][4] = {};

    // staging index precompute
    const int arow = tid >> 2, acol = (tid & 3) * 16;   // A: 16 elems/thread
    const int brow = tid >> 2, bcol = (tid & 3) * 64;   // B: 64 elems/thread (128B)
    const int agr   = m0 + arow;
    const bool aok  = agr < M;
    const int agclamp = aok ? agr : (M - 1);            // always-valid row

    const unsigned bs_lds = (unsigned)(size_t)(Bs + brow * BSTR2 + bcol);

    const int nkt = Kd / TKS;
    for (int kt = 0; kt < nkt; ++kt) {
        const int kb = kt * TKS;

        // ---- B tile 64x256 bf16: async DMA global -> LDS (8 x B128/thread) --
        {
            const __bf16* Bg = Bm + (size_t)(kb + brow) * ldb + n0 + bcol;
#pragma unroll
            for (int i = 0; i < 8; ++i)
                async_copy_b128(bs_lds + i * 16, Bg + i * 8);
        }
        // ---- A tile 64x64 (fp32 -> bf16), branch-free OOB zeroing ----------
        {
            const float* Ag = A + (size_t)agclamp * lda + kb + acol;
            float av[16];
#pragma unroll
            for (int i = 0; i < 16; ++i) av[i] = Ag[i];
#pragma unroll
            for (int i = 0; i < 16; ++i) av[i] = aok ? av[i] : 0.0f;
            __bf16* as = As + arow * ASTR + acol;
#pragma unroll
            for (int i = 0; i < 16; ++i) as[i] = (__bf16)av[i];
        }
        wait_async0();
        __syncthreads();

#pragma unroll
        for (int ks = 0; ks < 2; ++ks) {
            const int ko = ks * 32;
            v16bf bf[4];
#pragma unroll
            for (int b = 0; b < 4; ++b)
                bf[b] = load_b_frag(Bs, ko, wn * 64 + b * 16, lane);
#pragma unroll
            for (int a = 0; a < 2; ++a) {
                v16bf af = load_a_frag(As, wm * 32 + a * 16, ko, lane);
#pragma unroll
                for (int b = 0; b < 4; ++b)
                    acc[a][b] = __builtin_amdgcn_wmma_f32_16x16x32_bf16(
                        false, af, false, bf[b], (short)0, acc[a][b], false, false);
            }
        }
        __syncthreads();
    }

    // store: C 16x16 f32 layout -> lane N=(lane&15), VGPR r -> M = r + (lane>>4)*8
#pragma unroll
    for (int a = 0; a < 2; ++a) {
#pragma unroll
        for (int b = 0; b < 4; ++b) {
            const int col = n0 + wn * 64 + b * 16 + (lane & 15);
            const float bv = bias ? bias[col] : 0.0f;
#pragma unroll
            for (int r = 0; r < 8; ++r) {
                const int row = m0 + wm * 32 + a * 16 + (lane >> 4) * 8 + r;
                if (row < M) {
                    float v = acc[a][b][r] + bv;
                    if (relu) v = fmaxf(v, 0.0f);
                    C[(long long)row * ldc + col] = v;
                }
            }
        }
    }
}

// ---------------- fp32 -> bf16 weight conversion ----------------------------
__global__ __launch_bounds__(256) void f32_to_bf16_kernel(
    const float* __restrict__ src, __bf16* __restrict__ dst, int n)
{
    const int i = (blockIdx.x * 256 + threadIdx.x) * 4;
    if (i + 3 < n) {
        const float4 v = *(const float4*)(src + i);
        dst[i] = (__bf16)v.x; dst[i + 1] = (__bf16)v.y;
        dst[i + 2] = (__bf16)v.z; dst[i + 3] = (__bf16)v.w;
    } else {
        for (int j = i; j < n; ++j) dst[j] = (__bf16)src[j];
    }
}

// ---------------- attention: one block per (b,h), 128 threads = queries -----
__global__ __launch_bounds__(128) void attention_kernel(
    const float* __restrict__ Q, const float* __restrict__ Kx,
    const float* __restrict__ Vx, float* __restrict__ ctx)
{
    __shared__ float ldsK[SS * DKH];   // 32 KB
    const int b = blockIdx.x / HH;
    const int h = blockIdx.x % HH;
    const int t = threadIdx.x;
    const float scale = 0.125f;        // 1/sqrt(64)

    for (int i = t; i < SS * DKH; i += 128) {
        int j = i >> 6, d = i & 63;
        ldsK[i] = Kx[(size_t)(b * SS + j) * DD + h * DKH + d];
    }
    __syncthreads();

    float qreg[DKH];
#pragma unroll
    for (int d = 0; d < DKH; ++d)
        qreg[d] = Q[(size_t)(b * SS + t) * DD + h * DKH + d];

    float mx = -1e30f, l = 0.0f;
    for (int j = 0; j < SS; ++j) {
        float s = 0.0f;
#pragma unroll
        for (int d = 0; d < DKH; ++d) s += qreg[d] * ldsK[j * DKH + d];
        s *= scale;
        float mn = fmaxf(mx, s);
        l = l * __expf(mx - mn) + __expf(s - mn);
        mx = mn;
    }

    float acc[DKH];
#pragma unroll
    for (int d = 0; d < DKH; ++d) acc[d] = 0.0f;
    for (int j = 0; j < SS; ++j) {
        float s = 0.0f;
#pragma unroll
        for (int d = 0; d < DKH; ++d) s += qreg[d] * ldsK[j * DKH + d];
        const float p = __expf(s * scale - mx);
        const float* vrow = Vx + (size_t)(b * SS + j) * DD + h * DKH;
#pragma unroll
        for (int d = 0; d < DKH; ++d) acc[d] += p * vrow[d];
    }
    const float inv = 1.0f / l;
    float* crow = ctx + (size_t)(b * SS + t) * DD + h * DKH;
#pragma unroll
    for (int d = 0; d < DKH; ++d) crow[d] = acc[d] * inv;
}

// ---------------- fused residual add + LayerNorm (row of 768) ---------------
__global__ __launch_bounds__(256) void add_ln_kernel(
    const float* __restrict__ X, const float* __restrict__ Y,
    const float* __restrict__ G, const float* __restrict__ Bt,
    float* __restrict__ O)
{
    __shared__ float red[256];
    const int row = blockIdx.x, tid = threadIdx.x;
    const float* xr = X + (size_t)row * DD;
    const float* yr = Y + (size_t)row * DD;

    float v[3];
    float s = 0.0f;
#pragma unroll
    for (int i = 0; i < 3; ++i) { int c = tid + i * 256; v[i] = xr[c] + yr[c]; s += v[i]; }
    red[tid] = s; __syncthreads();
    for (int o = 128; o > 0; o >>= 1) { if (tid < o) red[tid] += red[tid + o]; __syncthreads(); }
    const float mu = red[0] * (1.0f / 768.0f);
    __syncthreads();
    float s2 = 0.0f;
#pragma unroll
    for (int i = 0; i < 3; ++i) { float d = v[i] - mu; s2 += d * d; }
    red[tid] = s2; __syncthreads();
    for (int o = 128; o > 0; o >>= 1) { if (tid < o) red[tid] += red[tid + o]; __syncthreads(); }
    const float rstd = rsqrtf(red[0] * (1.0f / 768.0f) + 1e-5f);
#pragma unroll
    for (int i = 0; i < 3; ++i) {
        int c = tid + i * 256;
        O[(size_t)row * DD + c] = (v[i] - mu) * rstd * G[c] + Bt[c];
    }
}

// ---------------- cat = [rmsnorm(emb[tok]), rmsnorm(h_prev)] ----------------
__global__ __launch_bounds__(256) void build_cat_kernel(
    const int* __restrict__ tok, const float* __restrict__ emb,
    const float* __restrict__ hprev, float* __restrict__ cat, int koff)
{
    __shared__ float red[256];
    const int r = blockIdx.x, tid = threadIdx.x;
    const int b = r / MM, m = r % MM;
    const int t = tok[b * SS + m + koff];
    const float* er = emb + (size_t)t * DD;
    const float* hr = hprev + (size_t)r * DD;

    float se = 0.0f, sh = 0.0f;
    for (int c = tid; c < DD; c += 256) { float a = er[c]; se += a * a; float d = hr[c]; sh += d * d; }
    red[tid] = se; __syncthreads();
    for (int o = 128; o > 0; o >>= 1) { if (tid < o) red[tid] += red[tid + o]; __syncthreads(); }
    const float ie = 1.0f / (sqrtf(red[0] * (1.0f / 768.0f)) + 1e-8f);
    __syncthreads();
    red[tid] = sh; __syncthreads();
    for (int o = 128; o > 0; o >>= 1) { if (tid < o) red[tid] += red[tid + o]; __syncthreads(); }
    const float ih = 1.0f / (sqrtf(red[0] * (1.0f / 768.0f)) + 1e-8f);

    for (int c = tid; c < DD; c += 256) {
        cat[(size_t)r * (2 * DD) + c]      = er[c] * ie;
        cat[(size_t)r * (2 * DD) + DD + c] = hr[c] * ih;
    }
}

// ---------------- gather h_prev (drop last K rows of each batch) ------------
__global__ __launch_bounds__(256) void gather_hprev_kernel(
    const float* __restrict__ hidden, float* __restrict__ hprev)
{
    const int i = blockIdx.x * 256 + threadIdx.x;
    if (i >= MROWS * DD) return;
    const int r = i / DD, c = i % DD;
    const int b = r / MM, m = r % MM;
    hprev[i] = hidden[(size_t)(b * SS + m) * DD + c];
}

// ---------------------------------------------------------------------------

static inline void gemm(const float* A, int lda, const __bf16* Bm, int ldb,
                        const float* bias, float* C, long long ldc,
                        int M, int N, int Kd, int relu, hipStream_t s)
{
    dim3 grid(N / TN, (M + TM - 1) / TM);
    gemm_bf16_wmma<<<grid, 256, 0, s>>>(A, lda, Bm, ldb, bias, C, ldc, M, Kd, relu);
}

static inline void convert_w(const float* src, __bf16* dst, size_t n, hipStream_t s)
{
    const unsigned nthr = (unsigned)(n / 4);
    f32_to_bf16_kernel<<<(nthr + 255) / 256, 256, 0, s>>>(src, dst, (int)n);
}

extern "C" void kernel_launch(void* const* d_in, const int* in_sizes, int n_in,
                              void* d_out, int out_size, void* d_ws, size_t ws_size,
                              hipStream_t stream)
{
    (void)in_sizes; (void)n_in; (void)out_size; (void)ws_size;

    // setup_inputs() order: x, input_tokens, then params in dict order
    const float* x      = (const float*)d_in[0];
    const int*   tokens = (const int*)  d_in[1];
    const float* wq = (const float*)d_in[2],  *bq = (const float*)d_in[3];
    const float* wk = (const float*)d_in[4],  *bk = (const float*)d_in[5];
    const float* wv = (const float*)d_in[6],  *bv = (const float*)d_in[7];
    const float* wo = (const float*)d_in[8],  *bo = (const float*)d_in[9];
    const float* w1 = (const float*)d_in[10], *b1 = (const float*)d_in[11];
    const float* w2 = (const float*)d_in[12], *b2 = (const float*)d_in[13];
    const float* ln1g = (const float*)d_in[14], *ln1b = (const float*)d_in[15];
    const float* ln2g = (const float*)d_in[16], *ln2b = (const float*)d_in[17];
    const float* wp = (const float*)d_in[18], *bp = (const float*)d_in[19];
    const float* wu = (const float*)d_in[20], *bu = (const float*)d_in[21];
    const float* emb = (const float*)d_in[22];
    float* out = (float*)d_out;

    // byte-granular bump allocator over d_ws (256B aligned blocks)
    char* base = (char*)d_ws;
    size_t off = 0;
    auto alloc = [&](size_t bytes) -> void* {
        void* p = base + off;
        off = (off + bytes + 255) & ~(size_t)255;
        return p;
    };

    const size_t SA = (size_t)BB * SS * DD;     // 512*768
    const size_t SB = (size_t)MROWS * DD;       // 496*768
    float* q    = (float*)alloc(SA * 4);
    float* kb_  = (float*)alloc(SA * 4);
    float* vb_  = (float*)alloc(SA * 4);
    float* ctx  = (float*)alloc(SA * 4);
    float* t0   = (float*)alloc(SA * 4);
    float* hbuf = (float*)alloc(SA * 4);
    float* f1   = (float*)alloc(SA * 4);
    float* f2   = (float*)alloc(SA * 4);
    float* hid  = (float*)alloc(SA * 4);
    float* hprev= (float*)alloc(SB * 4);
    float* cat  = (float*)alloc((size_t)MROWS * 2 * DD * 4);
    float* chA  = (float*)alloc(SB * 4);
    float* chB  = (float*)alloc(SB * 4);
    float* t1   = (float*)alloc(SB * 4);
    float* t2   = (float*)alloc(SB * 4);
    float* h1   = (float*)alloc(SB * 4);
    float* res  = (float*)alloc(SB * 4);

    const size_t WSQ = (size_t)DD * DD;          // 768*768
    const size_t WSP = (size_t)2 * DD * DD;      // 1536*768 (per head)
    const size_t WSU = (size_t)DD * VV;          // 768*32000
    __bf16* wqb = (__bf16*)alloc(WSQ * 2);
    __bf16* wkb = (__bf16*)alloc(WSQ * 2);
    __bf16* wvb = (__bf16*)alloc(WSQ * 2);
    __bf16* wob = (__bf16*)alloc(WSQ * 2);
    __bf16* w1b = (__bf16*)alloc(WSQ * 2);
    __bf16* w2b = (__bf16*)alloc(WSQ * 2);
    __bf16* wpb = (__bf16*)alloc(KK * WSP * 2);
    __bf16* wub = (__bf16*)alloc(WSU * 2);

    // ---- weight conversion (once per launch) ----
    convert_w(wq, wqb, WSQ, stream);
    convert_w(wk, wkb, WSQ, stream);
    convert_w(wv, wvb, WSQ, stream);
    convert_w(wo, wob, WSQ, stream);
    convert_w(w1, w1b, WSQ, stream);
    convert_w(w2, w2b, WSQ, stream);
    convert_w(wp, wpb, KK * WSP, stream);
    convert_w(wu, wub, WSU, stream);

    const int Mfull = BB * SS;   // 512

    // ---- stage A: full transformer layer ----
    gemm(x, DD, wqb, DD, bq, q,   DD, Mfull, DD, DD, 0, stream);
    gemm(x, DD, wkb, DD, bk, kb_, DD, Mfull, DD, DD, 0, stream);
    gemm(x, DD, wvb, DD, bv, vb_, DD, Mfull, DD, DD, 0, stream);
    attention_kernel<<<BB * HH, 128, 0, stream>>>(q, kb_, vb_, ctx);
    gemm(ctx, DD, wob, DD, bo, t0, DD, Mfull, DD, DD, 0, stream);
    add_ln_kernel<<<Mfull, 256, 0, stream>>>(x, t0, ln1g, ln1b, hbuf);
    gemm(hbuf, DD, w1b, DD, b1, f1, DD, Mfull, DD, DD, 1, stream);
    gemm(f1,   DD, w2b, DD, b2, f2, DD, Mfull, DD, DD, 0, stream);
    add_ln_kernel<<<Mfull, 256, 0, stream>>>(hbuf, f2, ln2g, ln2b, hid);

    // ---- stage B: K MTP heads ----
    gather_hprev_kernel<<<(MROWS * DD + 255) / 256, 256, 0, stream>>>(hid, hprev);
    const float* hp = hprev;
    for (int k = 0; k < KK; ++k) {
        float* ch = (k & 1) ? chB : chA;
        build_cat_kernel<<<MROWS, 256, 0, stream>>>(tokens, emb, hp, cat, 1 + k);
        gemm(cat, 2 * DD, wpb + (size_t)k * WSP, DD, bp + (size_t)k * DD,
             ch, DD, MROWS, DD, 2 * DD, 0, stream);
        // single_token_layer
        gemm(ch, DD, wvb, DD, bv, t1, DD, MROWS, DD, DD, 0, stream);
        gemm(t1, DD, wob, DD, bo, t2, DD, MROWS, DD, DD, 0, stream);
        add_ln_kernel<<<MROWS, 256, 0, stream>>>(ch, t2, ln1g, ln1b, h1);
        gemm(h1, DD, w1b, DD, b1, f1, DD, MROWS, DD, DD, 1, stream);
        gemm(f1, DD, w2b, DD, b2, f2, DD, MROWS, DD, DD, 0, stream);
        add_ln_kernel<<<MROWS, 256, 0, stream>>>(h1, f2, ln2g, ln2b, res);
        // logits -> out[b, m, k, :], row stride K*V, base offset k*V
        gemm(res, DD, wub, VV, bu, out + (size_t)k * VV, (long long)KK * VV,
             MROWS, VV, DD, 0, stream);
        hp = ch;
    }
}
